// DLRM_Net_65515431133905
// MI455X (gfx1250) — compile-verified, matrix operand order checked by reading, not run on hardware
//
#include <hip/hip_runtime.h>
#include <hip/hip_bf16.h>

typedef __attribute__((ext_vector_type(16))) _Float16 v16h;
typedef __attribute__((ext_vector_type(8)))  _Float16 h8;
typedef __attribute__((ext_vector_type(8)))  float    v8f;
typedef __attribute__((ext_vector_type(4)))  unsigned int u32x4;
typedef __attribute__((ext_vector_type(4)))  int i32x4;
typedef __attribute__((ext_vector_type(8)))  int i32x8;

#define B_SZ   8192
#define NTAB   26
#define VOCABN 100000
#define MSPA   64

// ---------------------------------------------------------------------------
// fp32 -> f16 conversion with K padding (rows x cols  ->  rows x ldd, zero pad)
// ---------------------------------------------------------------------------
__global__ void cvt_pad_f16(const float* __restrict__ src, _Float16* __restrict__ dst,
                            int rows, int cols, int ldd) {
  int i = blockIdx.x * blockDim.x + threadIdx.x;
  if (i >= rows * ldd) return;
  int r = i / ldd;
  int c = i - r * ldd;
  dst[i] = (c < cols) ? (_Float16)src[(size_t)r * cols + c] : (_Float16)0.f;
}

// ---------------------------------------------------------------------------
// Issue a TDM 2D tile load:  LDS[lds_off ..] = W[row0 .. row0+15][0..K-1]
// D# built per CDNA5 ISA 8.3/8.4: group0 = {count/type/addr}, group1 =
// {data_size=2B, tensor=[K,N], tile=[K,16], stride0=K}, groups 2/3 zero (2D).
// ---------------------------------------------------------------------------
__device__ __forceinline__ void tdm_load_strip(const _Float16* gsrc, unsigned lds_off,
                                               int K, int N) {
  unsigned long long ga = (unsigned long long)(uintptr_t)(const void*)gsrc;
  u32x4 g0;
  g0[0] = 1u;                                        // count=1, user descriptor
  g0[1] = lds_off;                                   // lds_addr (bytes)
  g0[2] = (unsigned)(ga & 0xFFFFFFFFu);              // global_addr[31:0]
  g0[3] = (unsigned)((ga >> 32) & 0x1FFFFFFu) | (2u << 30);  // addr[56:32], type=2
  i32x8 g1;
  g1[0] = (1 << 16);                                 // data_size = 1 -> 2 bytes
  g1[1] = (K & 0xFFFF) << 16;                        // tensor_dim0 = K (lo16)
  g1[2] = ((K >> 16) & 0xFFFF) | ((N & 0xFFFF) << 16); // dim0 hi16 | tensor_dim1 lo16
  g1[3] = ((N >> 16) & 0xFFFF) | ((K & 0xFFFF) << 16); // dim1 hi16 | tile_dim0 = K
  g1[4] = 16;                                        // tile_dim1 = 16, tile_dim2 = 0
  g1[5] = K;                                         // tensor_dim0_stride[31:0] = K
  g1[6] = 0;                                         // stride0 hi | stride1 lo
  g1[7] = 0;
  i32x4 gz;
  gz[0] = 0; gz[1] = 0; gz[2] = 0; gz[3] = 0;
#if __clang_major__ >= 23
  i32x8 g4;
  g4[0] = 0; g4[1] = 0; g4[2] = 0; g4[3] = 0; g4[4] = 0; g4[5] = 0; g4[6] = 0; g4[7] = 0;
  __builtin_amdgcn_tensor_load_to_lds(g0, g1, gz, gz, g4, 0);
#else
  __builtin_amdgcn_tensor_load_to_lds(g0, g1, gz, gz, 0);
#endif
}

// ---------------------------------------------------------------------------
// WMMA GEMM:  out[M,N] = act( A[M,K] * W[N,K]^T + bias[N] )
// Block = 256 threads = 8 waves = 128x16 macro-tile (wave w owns rows
// [128*blk + 16w, +16), all waves share column tile nt).  The shared 16xK
// weight strip is DMA'd into LDS once per block by the Tensor Data Mover
// (TENSORcnt), fragments are then read with ds_load_b128; per-wave A
// fragments stream from global with prefetch.  K, lda, ldw multiples of 32.
// ---------------------------------------------------------------------------
__global__ void gemm_wmma_tdm(const _Float16* __restrict__ A, int lda,
                              const _Float16* __restrict__ W, int ldw,
                              const float* __restrict__ bias,
                              _Float16* __restrict__ out, int ldo,
                              int N, int K, int do_relu) {
  __shared__ _Float16 ldsW[16 * 512];               // 16 rows x K (K <= 512)

  const int wid    = threadIdx.x >> 5;
  const int lane   = threadIdx.x & 31;
  const int tilesN = N >> 4;
  const int mblk   = blockIdx.x / tilesN;           // 128-row block index
  const int nt     = blockIdx.x - mblk * tilesN;    // 16-col tile index

  // --- TDM: one async strip load per block, issued by wave 0 only
  if (wid == 0) {
    tdm_load_strip(W + (size_t)nt * 16 * ldw, (unsigned)(uintptr_t)&ldsW[0], K, N);
    __builtin_amdgcn_s_wait_tensorcnt(0);
  }
  __syncthreads();

  const int    l15  = lane & 15;
  const int    koff = (lane >> 4) << 3;             // 0 or 8 (CDNA5 16-bit frag)
  const size_t arow = (size_t)(mblk * 128 + wid * 16 + l15) * lda;
  const _Float16* lrow = ldsW + (size_t)l15 * K;

  const float bn = bias[nt * 16 + l15];
  v8f c;
#pragma unroll
  for (int i = 0; i < 8; ++i) c[i] = bn;

  for (int k0 = 0; k0 < K; k0 += 32) {
    const _Float16* ap = A + arow + k0 + koff;      // global, pipelined + prefetch
    const _Float16* wp = lrow + k0 + koff;          // LDS (ds_load_b128)
    __builtin_prefetch(ap + 64, 0, 1);
    h8 a0 = *(const h8*)ap;
    h8 a1 = *(const h8*)(ap + 16);
    h8 w0 = *(const h8*)wp;
    h8 w1 = *(const h8*)(wp + 16);
    v16h af = __builtin_shufflevector(a0, a1, 0,1,2,3,4,5,6,7,8,9,10,11,12,13,14,15);
    v16h wf = __builtin_shufflevector(w0, w1, 0,1,2,3,4,5,6,7,8,9,10,11,12,13,14,15);
    c = __builtin_amdgcn_wmma_f32_16x16x32_f16(false, af, false, wf,
                                               (short)0, c, false, false);
  }

  // D layout: VGPR r -> M = r (lanes 0-15) / r+8 (lanes 16-31), N = lane&15
  const int m0 = mblk * 128 + wid * 16 + ((lane >> 4) << 3);
  const int n  = nt * 16 + l15;
#pragma unroll
  for (int r = 0; r < 8; ++r) {
    float v = c[r];
    if (do_relu) v = v > 0.f ? v : 0.f;
    out[(size_t)(m0 + r) * ldo + n] = (_Float16)v;
  }
}

// ---------------------------------------------------------------------------
// Embedding gather: Tbuf[b][1+t][0..63] = f16(Wemb[t][lS_i[t][b]][:])
// ---------------------------------------------------------------------------
__global__ void gather_emb(const float* __restrict__ Wemb, const int* __restrict__ lSi,
                           _Float16* __restrict__ Tbuf) {
  const int wid  = threadIdx.x >> 5;
  const int lane = threadIdx.x & 31;
  const int p    = blockIdx.x * 8 + wid;        // [0, 26*8192)
  const int t    = p >> 13;                     // /8192
  const int b    = p & (B_SZ - 1);
  const int idx  = lSi[(size_t)t * B_SZ + b];
  const float*    src = Wemb + ((size_t)t * VOCABN + (size_t)idx) * MSPA;
  _Float16*       dst = Tbuf + (size_t)b * (32 * 64) + (size_t)(1 + t) * 64;
  dst[lane]      = (_Float16)src[lane];
  dst[lane + 32] = (_Float16)src[lane + 32];
}

// ---------------------------------------------------------------------------
// Scatter bottom-MLP output x into Tbuf row 0 and R[:, :64]; zero pads.
// ---------------------------------------------------------------------------
__global__ void scatter_x(const _Float16* __restrict__ xh,
                          _Float16* __restrict__ Tbuf, _Float16* __restrict__ R) {
  int i = blockIdx.x * blockDim.x + threadIdx.x;   // B*64 threads
  int b = i >> 6;
  int c = i & 63;
  _Float16 v = xh[i];
  Tbuf[(size_t)b * (32 * 64) + c] = v;
  R[(size_t)b * 416 + c]          = v;
#pragma unroll
  for (int j = 0; j < 5; ++j)                      // zero pad rows 27..31
    Tbuf[(size_t)b * (32 * 64) + (size_t)(27 + j) * 64 + c] = (_Float16)0.f;
  if (c == 0) R[(size_t)b * 416 + 415] = (_Float16)0.f;   // zero pad col 415
}

// ---------------------------------------------------------------------------
// Interaction: per sample, Z = T*T^T for T[27x64] (padded 32x64); lower
// triangle scattered into R[b, 64 + m(m-1)/2 + n].  One wave per sample,
// three WMMA tiles (tile (0,1) is pure upper-triangle -> skipped).
// ---------------------------------------------------------------------------
__global__ void interact_wmma(const _Float16* __restrict__ Tbuf,
                              _Float16* __restrict__ R) {
  const int wid  = threadIdx.x >> 5;
  const int lane = threadIdx.x & 31;
  const int b    = blockIdx.x * 8 + wid;
  const _Float16* Tb = Tbuf + (size_t)b * (32 * 64);
  const int l15  = lane & 15;
  const int koff = (lane >> 4) << 3;

  v8f c00, c10, c11;
#pragma unroll
  for (int i = 0; i < 8; ++i) { c00[i] = 0.f; c10[i] = 0.f; c11[i] = 0.f; }

#pragma unroll
  for (int k0 = 0; k0 < 64; k0 += 32) {
    const _Float16* p0 = Tb + (size_t)l15 * 64 + k0 + koff;         // rows 0..15
    const _Float16* p1 = Tb + (size_t)(16 + l15) * 64 + k0 + koff;  // rows 16..31
    h8 x0 = *(const h8*)p0, x1 = *(const h8*)(p0 + 16);
    h8 y0 = *(const h8*)p1, y1 = *(const h8*)(p1 + 16);
    v16h f0 = __builtin_shufflevector(x0, x1, 0,1,2,3,4,5,6,7,8,9,10,11,12,13,14,15);
    v16h f1 = __builtin_shufflevector(y0, y1, 0,1,2,3,4,5,6,7,8,9,10,11,12,13,14,15);
    c00 = __builtin_amdgcn_wmma_f32_16x16x32_f16(false, f0, false, f0, (short)0, c00, false, false);
    c10 = __builtin_amdgcn_wmma_f32_16x16x32_f16(false, f1, false, f0, (short)0, c10, false, false);
    c11 = __builtin_amdgcn_wmma_f32_16x16x32_f16(false, f1, false, f1, (short)0, c11, false, false);
  }

  _Float16* Rb = R + (size_t)b * 416 + 64;
  const int mo = (lane >> 4) << 3;
#pragma unroll
  for (int r = 0; r < 8; ++r) {
    int m = mo + r, n = l15;                               // tile (0,0)
    if (n < m)                    Rb[m * (m - 1) / 2 + n] = (_Float16)c00[r];
    m = 16 + mo + r; n = l15;                              // tile (1,0): n<m always
    if (m < 27)                   Rb[m * (m - 1) / 2 + n] = (_Float16)c10[r];
    m = 16 + mo + r; n = 16 + l15;                         // tile (1,1)
    if (m < 27 && n < m)          Rb[m * (m - 1) / 2 + n] = (_Float16)c11[r];
  }
}

// ---------------------------------------------------------------------------
// Final layer: p[b] = sigmoid( z2[b,:256] . tW2 + tb2 )
// ---------------------------------------------------------------------------
__global__ void top_final(const _Float16* __restrict__ z2, const float* __restrict__ tW2,
                          const float* __restrict__ tb2, float* __restrict__ out) {
  int b = blockIdx.x * blockDim.x + threadIdx.x;
  if (b >= B_SZ) return;
  float acc = tb2[0];
  const _Float16* zr = z2 + (size_t)b * 256;
  for (int k = 0; k < 256; ++k) acc += (float)zr[k] * tW2[k];
  out[b] = 1.0f / (1.0f + __expf(-acc));
}

// ---------------------------------------------------------------------------
extern "C" void kernel_launch(void* const* d_in, const int* in_sizes, int n_in,
                              void* d_out, int out_size, void* d_ws, size_t ws_size,
                              hipStream_t stream) {
  (void)in_sizes; (void)n_in; (void)out_size; (void)ws_size;

  const float* dense_x = (const float*)d_in[0];
  const int*   lS_i    = (const int*)d_in[1];
  // d_in[2] = lS_o (arange tile, unused)
  const float* Wemb = (const float*)d_in[3];
  const float* bW0  = (const float*)d_in[4];   const float* bb0 = (const float*)d_in[5];
  const float* bW1  = (const float*)d_in[6];   const float* bb1 = (const float*)d_in[7];
  const float* bW2  = (const float*)d_in[8];   const float* bb2 = (const float*)d_in[9];
  const float* tW0  = (const float*)d_in[10];  const float* tb0 = (const float*)d_in[11];
  const float* tW1  = (const float*)d_in[12];  const float* tb1 = (const float*)d_in[13];
  const float* tW2  = (const float*)d_in[14];  const float* tb2 = (const float*)d_in[15];
  float* out = (float*)d_out;

  // workspace carve-out (all f16 buffers, 256B aligned)
  size_t off = 0;
  auto carve = [&](size_t bytes) {
    void* p = (char*)d_ws + off;
    off += (bytes + 255) & ~(size_t)255;
    return p;
  };
  _Float16* xin = (_Float16*)carve((size_t)B_SZ * 32 * 2);        // dense padded 13->32
  _Float16* wb0 = (_Float16*)carve((size_t)512 * 32 * 2);
  _Float16* wb1 = (_Float16*)carve((size_t)256 * 512 * 2);
  _Float16* wb2 = (_Float16*)carve((size_t)64 * 256 * 2);
  _Float16* wt0 = (_Float16*)carve((size_t)512 * 416 * 2);        // 415->416
  _Float16* wt1 = (_Float16*)carve((size_t)256 * 512 * 2);
  _Float16* h1  = (_Float16*)carve((size_t)B_SZ * 512 * 2);       // reused as z1
  _Float16* h2  = (_Float16*)carve((size_t)B_SZ * 256 * 2);       // reused as z2
  _Float16* xh  = (_Float16*)carve((size_t)B_SZ * 64 * 2);
  _Float16* Tb  = (_Float16*)carve((size_t)B_SZ * 32 * 64 * 2);   // padded 27->32
  _Float16* R   = (_Float16*)carve((size_t)B_SZ * 416 * 2);

  // 1) fp32 -> f16 conversions (with K padding)
  cvt_pad_f16<<<(B_SZ * 32 + 255) / 256, 256, 0, stream>>>(dense_x, xin, B_SZ, 13, 32);
  cvt_pad_f16<<<(512 * 32 + 255) / 256, 256, 0, stream>>>(bW0, wb0, 512, 13, 32);
  cvt_pad_f16<<<(256 * 512 + 255) / 256, 256, 0, stream>>>(bW1, wb1, 256, 512, 512);
  cvt_pad_f16<<<(64 * 256 + 255) / 256, 256, 0, stream>>>(bW2, wb2, 64, 256, 256);
  cvt_pad_f16<<<(512 * 416 + 255) / 256, 256, 0, stream>>>(tW0, wt0, 512, 415, 416);
  cvt_pad_f16<<<(256 * 512 + 255) / 256, 256, 0, stream>>>(tW1, wt1, 256, 512, 512);

  // 2) bottom MLP: 13(32)->512->256->64, ReLU each (TDM + WMMA)
  gemm_wmma_tdm<<<64 * 32, 256, 0, stream>>>(xin, 32, wb0, 32, bb0, h1, 512, 512, 32, 1);
  gemm_wmma_tdm<<<64 * 16, 256, 0, stream>>>(h1, 512, wb1, 512, bb1, h2, 256, 256, 512, 1);
  gemm_wmma_tdm<<<64 * 4,  256, 0, stream>>>(h2, 256, wb2, 256, bb2, xh, 64, 64, 256, 1);

  // 3) assemble T = [x ; 26 embeddings] (f16), pad rows zeroed
  scatter_x<<<(B_SZ * 64) / 256, 256, 0, stream>>>(xh, Tb, R);
  gather_emb<<<(NTAB * B_SZ) / 8, 256, 0, stream>>>(Wemb, lS_i, Tb);

  // 4) pairwise interaction Z = T T^T, lower triangle -> R[:,64:415] (WMMA)
  interact_wmma<<<B_SZ / 8, 256, 0, stream>>>(Tb, R);

  // 5) top MLP: 415(416)->512->256 (TDM + WMMA), then ->1 + sigmoid
  gemm_wmma_tdm<<<64 * 32, 256, 0, stream>>>(R, 416, wt0, 416, tb0, h1, 512, 512, 416, 1);
  gemm_wmma_tdm<<<64 * 16, 256, 0, stream>>>(h1, 512, wt1, 512, tb1, h2, 256, 256, 512, 1);
  top_final<<<(B_SZ + 255) / 256, 256, 0, stream>>>(h2, tW2, tb2, out);
}
